// RaNAMLPKernel_67645734912040
// MI455X (gfx1250) — matrix operand order repaired and checked by reading, not compiled, not run on hardware
//
#include <hip/hip_runtime.h>

typedef __attribute__((ext_vector_type(16))) _Float16 v16h;
typedef __attribute__((ext_vector_type(8)))  float    v8f;
typedef __attribute__((ext_vector_type(4)))  _Float16 h4;
typedef __attribute__((ext_vector_type(8)))  _Float16 h8;

#define BK      32
#define LDS_K   (BK + 8)      // padded stride in halves; 40*2=80B keeps 16B alignment

// ---------------------------------------------------------------------------
// Register-staged A tile: fetch (global -> VGPR, no use) then commit
// (VGPR -> cvt -> LDS) one K-step later so global latency hides behind WMMA.
// A is row-major [M][K]; LDS tile is [ROWS][BK] f16.
// ---------------------------------------------------------------------------
template<int ROWS, typename AT> struct ARegs;
template<int ROWS> struct ARegs<ROWS, float>    { float4 v[ROWS / 32]; };
template<int ROWS> struct ARegs<ROWS, _Float16> { h4     v[ROWS / 32]; };

template<int ROWS>
static __device__ __forceinline__ void fetch_a(ARegs<ROWS, float>& r,
                                               const float* __restrict__ A,
                                               int lda, int m0, int k0, int tid) {
    int r0 = tid >> 3;            // 0..31
    int c  = (tid & 7) * 4;       // 0,4,...,28
#pragma unroll
    for (int p = 0; p < ROWS / 32; ++p)
        r.v[p] = *(const float4*)(A + (size_t)(m0 + r0 + 32 * p) * lda + k0 + c);
}

template<int ROWS>
static __device__ __forceinline__ void fetch_a(ARegs<ROWS, _Float16>& r,
                                               const _Float16* __restrict__ A,
                                               int lda, int m0, int k0, int tid) {
    int r0 = tid >> 3;
    int c  = (tid & 7) * 4;
#pragma unroll
    for (int p = 0; p < ROWS / 32; ++p)
        r.v[p] = *(const h4*)(A + (size_t)(m0 + r0 + 32 * p) * lda + k0 + c);
}

template<int ROWS>
static __device__ __forceinline__ void commit_a(_Float16* __restrict__ As,
                                                const ARegs<ROWS, float>& r, int tid) {
    int r0 = tid >> 3;
    int c  = (tid & 7) * 4;
#pragma unroll
    for (int p = 0; p < ROWS / 32; ++p) {
        float4 v = r.v[p];
        h4 t = { (_Float16)v.x, (_Float16)v.y, (_Float16)v.z, (_Float16)v.w };
        *(h4*)(As + (r0 + 32 * p) * LDS_K + c) = t;
    }
}

template<int ROWS>
static __device__ __forceinline__ void commit_a(_Float16* __restrict__ As,
                                                const ARegs<ROWS, _Float16>& r, int tid) {
    int r0 = tid >> 3;
    int c  = (tid & 7) * 4;
#pragma unroll
    for (int p = 0; p < ROWS / 32; ++p)
        *(h4*)(As + (r0 + 32 * p) * LDS_K + c) = r.v[p];
}

// ---------------------------------------------------------------------------
// Register-staged B tile with transpose: B row-major [K][N] -> LDS [N][BK].
// Each thread owns a column segment; per-k global reads coalesce across lanes
// (consecutive lanes -> consecutive n); commit writes contiguous halves
// (vectorized ds stores).
// ---------------------------------------------------------------------------
template<int COLS> struct BRegs { float v[BK * COLS / 256]; };  // KPT floats

template<int COLS>
static __device__ __forceinline__ void fetch_b(BRegs<COLS>& r,
                                               const float* __restrict__ B,
                                               int ldb, int k0, int n0, int tid) {
    constexpr int TPC = 256 / COLS;
    constexpr int KPT = BK / TPC;
    int n  = tid & (COLS - 1);
    int kh = (tid / COLS) * KPT;
    const float* g = B + (size_t)(k0 + kh) * ldb + n0 + n;
#pragma unroll
    for (int kk = 0; kk < KPT; ++kk)
        r.v[kk] = g[(size_t)kk * ldb];
}

template<int COLS>
static __device__ __forceinline__ void commit_b(_Float16* __restrict__ Bs,
                                                const BRegs<COLS>& r, int tid) {
    constexpr int TPC = 256 / COLS;
    constexpr int KPT = BK / TPC;
    int n  = tid & (COLS - 1);
    int kh = (tid / COLS) * KPT;
    __align__(16) _Float16 tmp[KPT];
#pragma unroll
    for (int kk = 0; kk < KPT; ++kk)
        tmp[kk] = (_Float16)r.v[kk];
    _Float16* s = Bs + n * LDS_K + kh;
#pragma unroll
    for (int c = 0; c < KPT / 8; ++c)
        ((h8*)s)[c] = ((const h8*)tmp)[c];
}

// ---------------------------------------------------------------------------
// WMMA fragment loads (16-bit A 16x32, B 32x16), contiguous k order: k = 16*(lane/16) + elem
// ---------------------------------------------------------------------------
static __device__ __forceinline__ v16h load_a_frag(const _Float16* __restrict__ As,
                                                   int mbase, int lane) {
    int row = mbase + (lane & 15);
    int kh  = (lane < 16) ? 0 : 16;    // lanes 0-15: K0-15; lanes 16-31: K16-31
    const _Float16* p = As + row * LDS_K + kh;
    v16h a;
    ((uint4*)&a)[0] = *(const uint4*)(p);        // elems 0-7  -> VGPR0-3
    ((uint4*)&a)[1] = *(const uint4*)(p + 8);    // elems 8-15 -> VGPR4-7
    return a;
}

static __device__ __forceinline__ v16h load_b_frag(const _Float16* __restrict__ Bs,
                                                   int nbase, int lane) {
    int n  = nbase + (lane & 15);
    int k0 = (lane < 16) ? 0 : 16;     // lanes 0-15: K0-15; lanes 16-31: K16-31
    const _Float16* p = Bs + n * LDS_K + k0;
    v16h b;
    ((uint4*)&b)[0] = *(const uint4*)(p);
    ((uint4*)&b)[1] = *(const uint4*)(p + 8);
    return b;
}

// ---------------------------------------------------------------------------
// Single GEMM: C[M,N] = A[M,K] * B[K,N]; block tile 128x128x32, 8 waves (2x4),
// wave tile 64x32 -> 4x2 WMMA subtiles. Ping-pong LDS + register prefetch:
// global loads for K-step k+1 issue before the WMMAs of step k, waits land in
// the next commit. EPI 0: mask(|c|*norm[n] > thr) f32 store. EPI 1: plain.
// ---------------------------------------------------------------------------
template<typename AT, int EPI>
__global__ __launch_bounds__(256)
void gemm128(const AT* __restrict__ A, const float* __restrict__ B,
             float* __restrict__ C, const float* __restrict__ norm,
             float thresh, int M, int N, int K) {
    __shared__ __align__(16) _Float16 As[2][128 * LDS_K];
    __shared__ __align__(16) _Float16 Bs[2][128 * LDS_K];

    const int tid  = threadIdx.x;
    const int lane = tid & 31;
    const int wave = tid >> 5;
    const int wrow = wave >> 2;        // 0..1  (M)
    const int wcol = wave & 3;         // 0..3  (N)
    const int m0 = blockIdx.y * 128;
    const int n0 = blockIdx.x * 128;

    v8f acc[4][2] = {};

    ARegs<128, AT> ra;
    BRegs<128>     rb;
    fetch_a<128>(ra, A, K, m0, 0, tid);
    fetch_b<128>(rb, B, N, 0, n0, tid);

    int cur = 0;
    for (int k0 = 0; k0 < K; k0 += BK) {
        commit_a<128>(As[cur], ra, tid);   // waits on loads issued last step
        commit_b<128>(Bs[cur], rb, tid);
        __syncthreads();

        if (k0 + BK < K) {                 // issue next tile's global loads now
            fetch_a<128>(ra, A, K, m0, k0 + BK, tid);
            fetch_b<128>(rb, B, N, k0 + BK, n0, tid);
        }

        v16h a[4], b[2];
#pragma unroll
        for (int i = 0; i < 4; ++i) a[i] = load_a_frag(As[cur], wrow * 64 + i * 16, lane);
#pragma unroll
        for (int j = 0; j < 2; ++j) b[j] = load_b_frag(Bs[cur], wcol * 32 + j * 16, lane);
#pragma unroll
        for (int i = 0; i < 4; ++i)
#pragma unroll
            for (int j = 0; j < 2; ++j)
                acc[i][j] = __builtin_amdgcn_wmma_f32_16x16x32_f16(
                    false, a[i], false, b[j], (short)0, acc[i][j], false, false);

        cur ^= 1;
    }

    // C layout: lane 0-15 -> M=v, lane 16-31 -> M=v+8; N = lane&15
    const int lm = (lane < 16) ? 0 : 8;
    const int ln = lane & 15;
#pragma unroll
    for (int i = 0; i < 4; ++i)
#pragma unroll
        for (int j = 0; j < 2; ++j) {
            int gm = m0 + wrow * 64 + i * 16 + lm;
            int gn = n0 + wcol * 32 + j * 16 + ln;
            float nrm = (EPI == 0) ? norm[gn] : 0.0f;
#pragma unroll
            for (int v = 0; v < 8; ++v) {
                float val = acc[i][j][v];
                if (EPI == 0) val = (fabsf(val) * nrm > thresh) ? val : 0.0f;
                C[(size_t)(gm + v) * N + gn] = val;
            }
        }
}

// ---------------------------------------------------------------------------
// Dual GEMM + fused SwiGLU + down-mask:
//   u = bx_up * A_t_up, g = bx_gate * A_t_gate
//   h = silu(g) * u ; h = (|h| * wnorm[n] > thr) ? h : 0 ; store f16
// Block tile 128x64x32, 8 waves (4x2), wave tile 32x32 -> 2x2 subtiles,
// two accumulator sets, ping-pong LDS + register prefetch.
// ---------------------------------------------------------------------------
__global__ __launch_bounds__(256)
void gemm_dual_swiglu(const float* __restrict__ Au, const float* __restrict__ Bu,
                      const float* __restrict__ Ag, const float* __restrict__ Bg,
                      _Float16* __restrict__ H, const float* __restrict__ wnorm,
                      float thresh, int M, int N, int K) {
    __shared__ __align__(16) _Float16 Asu[2][128 * LDS_K];
    __shared__ __align__(16) _Float16 Asg[2][128 * LDS_K];
    __shared__ __align__(16) _Float16 Bsu[2][64 * LDS_K];
    __shared__ __align__(16) _Float16 Bsg[2][64 * LDS_K];

    const int tid  = threadIdx.x;
    const int lane = tid & 31;
    const int wave = tid >> 5;
    const int wrow = wave >> 1;        // 0..3 (M)
    const int wcol = wave & 1;         // 0..1 (N)
    const int m0 = blockIdx.y * 128;
    const int n0 = blockIdx.x * 64;

    v8f accu[2][2] = {};
    v8f accg[2][2] = {};

    ARegs<128, float> rau, rag;
    BRegs<64>         rbu, rbg;
    fetch_a<128>(rau, Au, K, m0, 0, tid);
    fetch_a<128>(rag, Ag, K, m0, 0, tid);
    fetch_b<64>(rbu, Bu, N, 0, n0, tid);
    fetch_b<64>(rbg, Bg, N, 0, n0, tid);

    int cur = 0;
    for (int k0 = 0; k0 < K; k0 += BK) {
        commit_a<128>(Asu[cur], rau, tid);
        commit_a<128>(Asg[cur], rag, tid);
        commit_b<64>(Bsu[cur], rbu, tid);
        commit_b<64>(Bsg[cur], rbg, tid);
        __syncthreads();

        if (k0 + BK < K) {
            fetch_a<128>(rau, Au, K, m0, k0 + BK, tid);
            fetch_a<128>(rag, Ag, K, m0, k0 + BK, tid);
            fetch_b<64>(rbu, Bu, N, k0 + BK, n0, tid);
            fetch_b<64>(rbg, Bg, N, k0 + BK, n0, tid);
        }

        v16h a[2], b[2];
#pragma unroll
        for (int i = 0; i < 2; ++i) a[i] = load_a_frag(Asu[cur], wrow * 32 + i * 16, lane);
#pragma unroll
        for (int j = 0; j < 2; ++j) b[j] = load_b_frag(Bsu[cur], wcol * 32 + j * 16, lane);
#pragma unroll
        for (int i = 0; i < 2; ++i)
#pragma unroll
            for (int j = 0; j < 2; ++j)
                accu[i][j] = __builtin_amdgcn_wmma_f32_16x16x32_f16(
                    false, a[i], false, b[j], (short)0, accu[i][j], false, false);

#pragma unroll
        for (int i = 0; i < 2; ++i) a[i] = load_a_frag(Asg[cur], wrow * 32 + i * 16, lane);
#pragma unroll
        for (int j = 0; j < 2; ++j) b[j] = load_b_frag(Bsg[cur], wcol * 32 + j * 16, lane);
#pragma unroll
        for (int i = 0; i < 2; ++i)
#pragma unroll
            for (int j = 0; j < 2; ++j)
                accg[i][j] = __builtin_amdgcn_wmma_f32_16x16x32_f16(
                    false, a[i], false, b[j], (short)0, accg[i][j], false, false);

        cur ^= 1;
    }

    const int lm = (lane < 16) ? 0 : 8;
    const int ln = lane & 15;
#pragma unroll
    for (int i = 0; i < 2; ++i)
#pragma unroll
        for (int j = 0; j < 2; ++j) {
            int gm = m0 + wrow * 32 + i * 16 + lm;
            int gn = n0 + wcol * 32 + j * 16 + ln;
            float nrm = wnorm[gn];
#pragma unroll
            for (int v = 0; v < 8; ++v) {
                float u = accu[i][j][v];
                float g = accg[i][j][v];
                // silu(g)*u with fast reciprocal (v_rcp_f32, not IEEE divide)
                float sig = __builtin_amdgcn_rcpf(1.0f + __expf(-g));
                float val = u * g * sig;
                val = (fabsf(val) * nrm > thresh) ? val : 0.0f;
                H[(size_t)(gm + v) * N + gn] = (_Float16)val;
            }
        }
}

// ---------------------------------------------------------------------------
// Host-side launcher
// ---------------------------------------------------------------------------
extern "C" void kernel_launch(void* const* d_in, const int* in_sizes, int n_in,
                              void* d_out, int out_size, void* d_ws, size_t ws_size,
                              hipStream_t stream) {
    constexpr int S = 4096, D = 4096, R = 1024, DFF = 11008;
    constexpr float THRESH_UP = 0.4f, THRESH_GATE = 0.4f, THRESH_DOWN = 0.1f;

    const float* x        = (const float*)d_in[0];   // [S, D]
    const float* B_up     = (const float*)d_in[1];   // [D, R]
    const float* A_t_up   = (const float*)d_in[2];   // [R, DFF]
    const float* B_gate   = (const float*)d_in[3];   // [D, R]
    const float* A_t_gate = (const float*)d_in[4];   // [R, DFF]
    const float* bn_up    = (const float*)d_in[5];   // [R]
    const float* bn_gate  = (const float*)d_in[6];   // [R]
    const float* wn_down  = (const float*)d_in[7];   // [DFF]
    const float* W_t_down = (const float*)d_in[8];   // [DFF, D]
    float* out = (float*)d_out;                      // [S, D]

    // workspace: bx_up f32 [S,R] | bx_gate f32 [S,R] | h f16 [S,DFF]  (~122 MB)
    float*    bxu = (float*)d_ws;
    float*    bxg = bxu + (size_t)S * R;
    _Float16* H   = (_Float16*)(bxg + (size_t)S * R);

    dim3 blk(256);

    // Stage 1: masked low-rank coefficients (two GEMMs, K = D)
    gemm128<float, 0><<<dim3(R / 128, S / 128), blk, 0, stream>>>(
        x, B_up, bxu, bn_up, THRESH_UP, S, R, D);
    gemm128<float, 0><<<dim3(R / 128, S / 128), blk, 0, stream>>>(
        x, B_gate, bxg, bn_gate, THRESH_GATE, S, R, D);

    // Stage 2: dual expansion + fused SwiGLU + down-mask, K = R
    gemm_dual_swiglu<<<dim3(DFF / 64, S / 128), blk, 0, stream>>>(
        bxu, A_t_up, bxg, A_t_gate, H, wn_down, THRESH_DOWN, S, DFF, R);

    // Stage 3: down projection to output, K = DFF
    gemm128<_Float16, 1><<<dim3(D / 128, S / 128), blk, 0, stream>>>(
        H, W_t_down, out, nullptr, 0.0f, S, D, DFF);
}
